// SVDQW4A4Linear_15934328668577
// MI455X (gfx1250) — 16-bit and 8-bit matrix operand order repaired and checked by reading, compiled, not run on hardware
//
#include <hip/hip_runtime.h>
#include <stdint.h>

#define T_TOK   16384
#define IN_F    2048
#define OUT_F   2048
#define RANK    32
#define GROUP   64
#define NGRP    (IN_F / GROUP)
#define MT      (T_TOK / 16)
#define NT      (OUT_F / 16)

typedef __attribute__((ext_vector_type(2)))  float  v2f;
typedef __attribute__((ext_vector_type(4)))  float  v4f;
typedef __attribute__((ext_vector_type(8)))  float  v8f;
typedef __attribute__((ext_vector_type(4)))  int    v4i;
typedef __attribute__((ext_vector_type(8)))  int    v8i;
typedef __attribute__((ext_vector_type(8)))  __bf16 v8bf;
typedef __attribute__((ext_vector_type(16))) __bf16 v16bf;

#define OFF_QA   ((size_t)0)
#define OFF_ASC  ((size_t)33554432)
#define OFF_QB   ((size_t)35651584)
#define OFF_T    ((size_t)39845888)
#define OFF_XSB  ((size_t)41943040)
#define OFF_PDB  ((size_t)109051904)
#define OFF_PUB  ((size_t)109182976)

__device__ __forceinline__ uint32_t f32_to_bf16(float f) {
  uint32_t u = __float_as_uint(f);
  return (u + 0x7fffu + ((u >> 16) & 1u)) >> 16;
}

__device__ __forceinline__ void async_g2l_b128(void* lds_dst, const void* gsrc) {
  uint32_t ldsa = (uint32_t)(uintptr_t)lds_dst;
  uint64_t ga   = (uint64_t)(uintptr_t)gsrc;
  asm volatile("global_load_async_to_lds_b128 %0, %1, off"
               :: "v"(ldsa), "v"(ga) : "memory");
}
__device__ __forceinline__ void wait_async_zero() {
  asm volatile("s_wait_asynccnt 0x0" ::: "memory");
}

__global__ void k_wprep(const int8_t* __restrict__ qw, uint8_t* __restrict__ qB) {
  int idx = blockIdx.x * 256 + threadIdx.x;
  int n = idx >> 10;
  int j = idx & 1023;
  int p  = (int)qw[idx];
  int lo = (p << 28) >> 28;
  int hi = p >> 4;
  int k  = 2 * j;
  int g  = k >> 6;
  int kk = k & 63;
  int ntile = n >> 4, nn = n & 15;
  int kr   = kk & 31;
  int half = (kr >> 4) & 1;
  int v    = ((kr & 15) >> 2) + ((kk >= 32) ? 4 : 0);
  int b    = kk & 3;
  int L    = nn + 16 * half;
  size_t off = ((size_t)(ntile * NGRP + g)) * 1024 + (size_t)L * 32 + v * 4 + b;
  qB[off]     = (uint8_t)(lo & 0xff);
  qB[off + 1] = (uint8_t)(hi & 0xff);
}

__global__ void k_pdprep(const float* __restrict__ pd, uint32_t* __restrict__ pdB) {
  int idx = blockIdx.x * 256 + threadIdx.x;
  int v  = idx & 7;
  int L  = (idx >> 3) & 31;
  int nt = (idx >> 8) & 1;
  int c  = idx >> 9;
  int r  = nt * 16 + (L & 15);
  int k  = c * 32 + 16 * (L >> 4) + 2 * v;
  uint32_t lo = f32_to_bf16(pd[(size_t)k * RANK + r]);
  uint32_t hi = f32_to_bf16(pd[(size_t)(k + 1) * RANK + r]);
  pdB[idx] = lo | (hi << 16);
}

__global__ void k_puprep(const float* __restrict__ pu, float* __restrict__ puB) {
  int idx = blockIdx.x * 256 + threadIdx.x;
  int v  = idx & 1;
  int L  = (idx >> 1) & 31;
  int c  = (idx >> 6) & 7;
  int nt = idx >> 9;
  int n  = nt * 16 + (L & 15);
  int r  = 4 * c + 2 * (L >> 4) + v;
  puB[idx] = pu[(size_t)n * RANK + r];
}

__global__ void __launch_bounds__(256) k_quant(
    const float* __restrict__ x, const float* __restrict__ smooth,
    uint8_t* __restrict__ qA, float* __restrict__ ascaleT,
    uint32_t* __restrict__ xsb) {
  int lane = threadIdx.x & 31;
  int wave = threadIdx.x >> 5;
  int t = blockIdx.x * 8 + wave;
  int g = lane;

  const float* xp = x + (size_t)t * IN_F + g * GROUP;
  const float* sp = smooth + g * GROUP;
  float xs[GROUP];
  #pragma unroll
  for (int j = 0; j < GROUP; j += 4) {
    v4f xv = __builtin_nontemporal_load((const v4f*)(xp + j));
    v4f sv = *(const v4f*)(sp + j);
    xs[j + 0] = xv.x / sv.x;
    xs[j + 1] = xv.y / sv.y;
    xs[j + 2] = xv.z / sv.z;
    xs[j + 3] = xv.w / sv.w;
  }

  uint32_t* xd = xsb + ((size_t)t * IN_F + (size_t)g * GROUP) / 2;
  #pragma unroll
  for (int j = 0; j < GROUP; j += 2)
    __builtin_nontemporal_store(f32_to_bf16(xs[j]) | (f32_to_bf16(xs[j + 1]) << 16),
                                xd + (j >> 1));

  float amax = 0.0f;
  #pragma unroll
  for (int j = 0; j < GROUP; ++j) amax = fmaxf(amax, fabsf(xs[j]));
  float ascale = (amax > 0.0f) ? (amax / 7.0f) : 1.0f;
  ascaleT[(size_t)g * T_TOK + t] = ascale;

  int q[GROUP];
  #pragma unroll
  for (int j = 0; j < GROUP; ++j) {
    float qf = rintf(xs[j] / ascale);
    qf = fminf(fmaxf(qf, -8.0f), 7.0f);
    q[j] = (int)qf;
  }

  int m = t & 15;
  int mtile = t >> 4;
  uint8_t* base = qA + ((size_t)(mtile * NGRP + g)) * 1024;
  #pragma unroll
  for (int h = 0; h < 2; ++h) {
    uint32_t dw[8];
    #pragma unroll
    for (int v = 0; v < 8; ++v) {
      int j0 = 32 * (v >> 2) + 16 * h + 4 * (v & 3);
      dw[v] = (uint32_t)(q[j0] & 255) | ((uint32_t)(q[j0 + 1] & 255) << 8) |
              ((uint32_t)(q[j0 + 2] & 255) << 16) | ((uint32_t)(q[j0 + 3] & 255) << 24);
    }
    uint32_t* dst = (uint32_t*)(base + (size_t)(m + 16 * h) * 32);
    v4i dlo = {(int)dw[0], (int)dw[1], (int)dw[2], (int)dw[3]};
    v4i dhi = {(int)dw[4], (int)dw[5], (int)dw[6], (int)dw[7]};
    *(v4i*)(dst)     = dlo;
    *(v4i*)(dst + 4) = dhi;
  }
}

__global__ void __launch_bounds__(256) k_lowrank(
    const uint32_t* __restrict__ xsb, const uint32_t* __restrict__ pdB,
    float* __restrict__ tmat) {
  int lane = threadIdx.x & 31;
  int wave = threadIdx.x >> 5;
  int mtile = blockIdx.x * 8 + wave;
  int h  = lane >> 4;
  int ln = lane & 15;
  const __bf16* xrow = (const __bf16*)xsb + (size_t)(mtile * 16 + ln) * IN_F;

  v8f acc0 = {}, acc1 = {};
  for (int c = 0; c < 64; ++c) {
    int k0 = c * 32 + 16 * h;
    v8bf alo = __builtin_nontemporal_load((const v8bf*)(xrow + k0));
    v8bf ahi = __builtin_nontemporal_load((const v8bf*)(xrow + k0 + 8));
    v16bf A = __builtin_shufflevector(alo, ahi, 0,1,2,3,4,5,6,7,8,9,10,11,12,13,14,15);

    const __bf16* bp0 = (const __bf16*)(pdB + ((size_t)(c * 2 + 0) * 32 + lane) * 8);
    const __bf16* bp1 = (const __bf16*)(pdB + ((size_t)(c * 2 + 1) * 32 + lane) * 8);
    v8bf b0l = *(const v8bf*)(bp0);  v8bf b0h = *(const v8bf*)(bp0 + 8);
    v8bf b1l = *(const v8bf*)(bp1);  v8bf b1h = *(const v8bf*)(bp1 + 8);
    v16bf B0 = __builtin_shufflevector(b0l, b0h, 0,1,2,3,4,5,6,7,8,9,10,11,12,13,14,15);
    v16bf B1 = __builtin_shufflevector(b1l, b1h, 0,1,2,3,4,5,6,7,8,9,10,11,12,13,14,15);

    acc0 = __builtin_amdgcn_wmma_f32_16x16x32_bf16(false, A, false, B0, (short)0, acc0, false, false);
    acc1 = __builtin_amdgcn_wmma_f32_16x16x32_bf16(false, A, false, B1, (short)0, acc1, false, false);
  }

  float* tp = tmat + (size_t)(mtile * 16 + 8 * h) * RANK + ln;
  #pragma unroll
  for (int v = 0; v < 8; ++v) {
    tp[(size_t)v * RANK]      = acc0[v];
    tp[(size_t)v * RANK + 16] = acc1[v];
  }
}

__global__ void __launch_bounds__(256) k_gemm(
    const uint8_t* __restrict__ qA, const uint8_t* __restrict__ qB,
    const float* __restrict__ ascaleT, const float* __restrict__ wscales,
    const float* __restrict__ tmat, const float* __restrict__ puB,
    const float* __restrict__ bias, float* __restrict__ out) {
  __shared__ uint8_t shA[NGRP * 1024];
  __shared__ float   shAs[NGRP * 16];
  int tid  = threadIdx.x;
  int lane = tid & 31;
  int wave = tid >> 5;
  int mtile = blockIdx.x >> 2;
  int nc    = (blockIdx.x & 3) * 8 + wave;
  int h  = lane >> 4;
  int ln = lane & 15;

  {
    const uint8_t* src = qA + (size_t)mtile * (NGRP * 1024);
    #pragma unroll
    for (int r = 0; r < 8; ++r) {
      int off = r * 4096 + tid * 16;
      async_g2l_b128(shA + off, src + off);
    }
    #pragma unroll
    for (int r = 0; r < 2; ++r) {
      int i = r * 256 + tid;
      shAs[i] = ascaleT[(size_t)(i >> 4) * T_TOK + mtile * 16 + (i & 15)];
    }
    wait_async_zero();
    __syncthreads();
  }

  v8f acc[4] = {{}, {}, {}, {}};
  v8i czero = {};
  const uint8_t* bChunk = qB + (size_t)(nc * 4) * NGRP * 1024 + (size_t)lane * 32;

  v4i b0[4], b1[4];
  float wsv[4];
  #pragma unroll
  for (int nt = 0; nt < 4; ++nt) {
    const uint8_t* bBase = bChunk + (size_t)nt * NGRP * 1024;
    b0[nt] = *(const v4i*)(bBase);
    b1[nt] = *(const v4i*)(bBase + 16);
    wsv[nt] = wscales[nc * 64 + nt * 16 + ln];
  }

  for (int g = 0; g < NGRP; ++g) {
    const v4i* ap = (const v4i*)(shA + g * 1024 + lane * 32);
    v4i a0 = ap[0], a1 = ap[1];
    v8i A = __builtin_shufflevector(a0, a1, 0, 1, 2, 3, 4, 5, 6, 7);
    v4f s0 = *(const v4f*)(shAs + g * 16 + 8 * h);
    v4f s1 = *(const v4f*)(shAs + g * 16 + 8 * h + 4);

    v4i nb0[4], nb1[4];
    float nws[4];
    if (g + 1 < NGRP) {
      #pragma unroll
      for (int nt = 0; nt < 4; ++nt) {
        const uint8_t* bBase = bChunk + (size_t)(nt * NGRP + (g + 1)) * 1024;
        nb0[nt] = *(const v4i*)(bBase);
        nb1[nt] = *(const v4i*)(bBase + 16);
        nws[nt] = wscales[(size_t)(g + 1) * OUT_F + nc * 64 + nt * 16 + ln];
      }
    }

    #pragma unroll
    for (int nt = 0; nt < 4; ++nt) {
      v8i B = __builtin_shufflevector(b0[nt], b1[nt], 0, 1, 2, 3, 4, 5, 6, 7);
      v8i d = __builtin_amdgcn_wmma_i32_16x16x64_iu8(true, A, true, B, czero, false, false);
      float ws = wsv[nt];
      #pragma unroll
      for (int v = 0; v < 8; ++v) {
        float sc = ((v < 4) ? s0[v] : s1[v - 4]) * ws;
        acc[nt][v] = __builtin_fmaf((float)d[v], sc, acc[nt][v]);
      }
    }

    if (g + 1 < NGRP) {
      #pragma unroll
      for (int nt = 0; nt < 4; ++nt) { b0[nt] = nb0[nt]; b1[nt] = nb1[nt]; wsv[nt] = nws[nt]; }
    }
  }

  const float* trow = tmat + (size_t)(mtile * 16 + ln) * RANK + 2 * h;
  v2f tA[8];
  #pragma unroll
  for (int c = 0; c < 8; ++c) tA[c] = *(const v2f*)(trow + 4 * c);

  #pragma unroll
  for (int nt = 0; nt < 4; ++nt) {
    const float* pub = puB + (size_t)(nc * 4 + nt) * 512 + lane * 2;
    #pragma unroll
    for (int c = 0; c < 8; ++c) {
      v2f Bp = *(const v2f*)(pub + c * 64);
      acc[nt] = __builtin_amdgcn_wmma_f32_16x16x4_f32(false, tA[c], false, Bp, (short)0, acc[nt], false, false);
    }
    float bv = bias[nc * 64 + nt * 16 + ln];
    float* orow = out + (size_t)(mtile * 16 + 8 * h) * OUT_F + nc * 64 + nt * 16 + ln;
    #pragma unroll
    for (int v = 0; v < 8; ++v)
      __builtin_nontemporal_store(acc[nt][v] + bv, orow + (size_t)v * OUT_F);
  }
}

extern "C" void kernel_launch(void* const* d_in, const int* in_sizes, int n_in,
                              void* d_out, int out_size, void* d_ws, size_t ws_size,
                              hipStream_t stream) {
  const float*  x      = (const float*)d_in[0];
  const int8_t* qw     = (const int8_t*)d_in[1];
  const float*  wsc    = (const float*)d_in[2];
  const float*  smooth = (const float*)d_in[3];
  const float*  bias   = (const float*)d_in[4];
  const float*  pd     = (const float*)d_in[5];
  const float*  pu     = (const float*)d_in[6];
  float* out = (float*)d_out;

  uint8_t*  ws   = (uint8_t*)d_ws;
  uint8_t*  qA   = ws + OFF_QA;
  float*    asT  = (float*)(ws + OFF_ASC);
  uint8_t*  qB   = ws + OFF_QB;
  float*    tmat = (float*)(ws + OFF_T);
  uint32_t* xsb  = (uint32_t*)(ws + OFF_XSB);
  uint32_t* pdB  = (uint32_t*)(ws + OFF_PDB);
  float*    puB  = (float*)(ws + OFF_PUB);

  k_wprep  <<<(OUT_F * (IN_F / 2)) / 256, 256, 0, stream>>>(qw, qB);
  k_pdprep <<<32768 / 256, 256, 0, stream>>>(pd, pdB);
  k_puprep <<<65536 / 256, 256, 0, stream>>>(pu, puB);
  k_quant  <<<T_TOK / 8, 256, 0, stream>>>(x, smooth, qA, asT, xsb);
  k_lowrank<<<MT / 8, 256, 0, stream>>>(xsb, pdB, tmat);
  k_gemm   <<<MT * 4, 256, 0, stream>>>(qA, qB, asT, wsc, tmat, puB, bias, out);
}
